// Encoder_15324443312436
// MI455X (gfx1250) — matrix operand order repaired and checked by reading, not compiled, not run on hardware
//
#include <hip/hip_runtime.h>

#define Bsz  64
#define Tlen 512
#define Hdim 1024
#define Lnum 4
#define NG   4096        // 4*H
#define KP   16          // k-steps (x32) per K-quarter: 4 waves x 512 elems

typedef __attribute__((ext_vector_type(16))) __bf16 v16bf;
typedef __attribute__((ext_vector_type(8)))  float  v8f;

union FragBF { v16bf v; uint4 u[2]; };

__device__ __forceinline__ unsigned short f2bf(float f) {
  unsigned int u = __float_as_uint(f);
  u += 0x7FFFu + ((u >> 16) & 1u);   // round-to-nearest-even
  return (unsigned short)(u >> 16);
}

__device__ __forceinline__ float sigmoidf(float x) {
  return 1.0f / (1.0f + __expf(-x));
}

// A tile: 16x32 bf16, row-major, base at (row0, k0) of tile, leading dim lda.
// ISA layout: lane<16 -> row=lane,    K in {0..7} U {16..23}
//             lane>=16 -> row=lane-16, K in {8..15} U {24..31}
__device__ __forceinline__ v16bf load_a_frag(const unsigned short* A, int lda) {
  int lane = threadIdx.x & 31;
  int row  = lane & 15;
  int kh   = (lane >> 4) << 3;              // 0 or 8
  FragBF f;
  const unsigned short* p = A + (size_t)row * lda + kh;
  f.u[0] = *(const uint4*)(p);              // K = kh .. kh+7
  f.u[1] = *(const uint4*)(p + 16);         // K = kh+16 .. kh+23
  return f.v;
}

// B tile: 32x16 (KxN). W stores B^T row-major (row n = output col, contiguous K).
// Layout: lane<16 -> col=lane, K in {0..7} U {16..23} ; lane>=16 -> col=lane-16, K in {8..15} U {24..31}
__device__ __forceinline__ v16bf load_b_frag(const unsigned short* W, int ldw) {
  int lane = threadIdx.x & 31;
  int col  = lane & 15;
  int kb   = (lane >> 4) << 3;              // 0 or 8
  FragBF f;
  const unsigned short* p = W + (size_t)col * ldw + kb;
  f.u[0] = *(const uint4*)(p);              // K = kb .. kb+7
  f.u[1] = *(const uint4*)(p + 16);         // K = kb+16 .. kb+23
  return f.v;
}

// One K-quarter pass: acc[g] += A(16x512) @ Wb[g]^T(16x512) for 4 gate tiles.
// 4 rotating fragment sets, prefetch distance 2, fully unrolled (KP=16).
// With 4 waves/block * 256 blocks, load stalls are hidden by wave switching.
__device__ __forceinline__ void gemm_pass(const unsigned short* __restrict__ A,
                                          const unsigned short* const (&Wb)[4],
                                          v8f (&acc)[4]) {
  v16bf aS[4];
  v16bf bS[4][4];

  #pragma unroll
  for (int p = 0; p < 2; ++p) {
    aS[p] = load_a_frag(A + p * 32, Hdim);
    #pragma unroll
    for (int g = 0; g < 4; ++g) bS[p][g] = load_b_frag(Wb[g] + p * 32, Hdim);
  }

  #pragma unroll
  for (int kk = 0; kk < KP; ++kk) {
    if (kk + 2 < KP) {
      const int s = (kk + 2) & 3;
      aS[s] = load_a_frag(A + (kk + 2) * 32, Hdim);
      #pragma unroll
      for (int g = 0; g < 4; ++g)
        bS[s][g] = load_b_frag(Wb[g] + (kk + 2) * 32, Hdim);
    }
    const int cu = kk & 3;
    #pragma unroll
    for (int g = 0; g < 4; ++g)
      acc[g] = __builtin_amdgcn_wmma_f32_16x16x32_bf16(
          false, aS[cu], false, bS[cu][g], (short)0, acc[g], false, false);
  }
}

// ---------------- setup kernels ----------------

__global__ void convert_weights_kernel(const float* __restrict__ Wih,
                                       const float* __restrict__ Whh,
                                       unsigned short* __restrict__ wih_bf,
                                       unsigned short* __restrict__ whh_bf, int n) {
  int idx = blockIdx.x * blockDim.x + threadIdx.x;
  if (idx < n) {
    wih_bf[idx] = f2bf(Wih[idx]);
    whh_bf[idx] = f2bf(Whh[idx]);
  }
}

__global__ void bias_sum_kernel(const float* __restrict__ bih,
                                const float* __restrict__ bhh,
                                float* __restrict__ bias, int n) {
  int idx = blockIdx.x * blockDim.x + threadIdx.x;
  if (idx < n) bias[idx] = bih[idx] + bhh[idx];
}

__global__ void init_state_kernel(const float* __restrict__ h0,
                                  const float* __restrict__ c0,
                                  float* __restrict__ h, float* __restrict__ c,
                                  unsigned short* __restrict__ hbf0, int n) {
  int idx = blockIdx.x * blockDim.x + threadIdx.x;
  if (idx < n) {
    float hv = h0[idx];
    h[idx] = hv; c[idx] = c0[idx]; hbf0[idx] = f2bf(hv);
  }
}

// lengths + stable descending argsort (B=64, one block)
__global__ void lengths_order_kernel(const int* __restrict__ x,
                                     int* __restrict__ len, int* __restrict__ order) {
  __shared__ int sl[Bsz];
  int b = threadIdx.x;
  int cnt = 0;
  for (int t = 0; t < Tlen; ++t) cnt += (x[(size_t)b * Tlen + t] > 0) ? 1 : 0;
  sl[b] = cnt; len[b] = cnt;
  __syncthreads();
  int r = 0;
  for (int j = 0; j < Bsz; ++j) {
    int lj = sl[j];
    if (lj > cnt || (lj == cnt && j < b)) ++r;
  }
  order[r] = b;   // stable argsort of -length
}

__global__ void embed_gather_kernel(const int* __restrict__ x,
                                    const float* __restrict__ emb,
                                    unsigned short* __restrict__ xbf, int t) {
  int idx = blockIdx.x * blockDim.x + threadIdx.x;  // B*H
  int b = idx >> 10;
  int j = idx & (Hdim - 1);
  int tok = x[(size_t)b * Tlen + t];
  xbf[idx] = f2bf(emb[(size_t)tok * Hdim + j]);
}

// ------- fused LSTM layer step: split-K WMMA GEMM + LDS reduce + pointwise -------
// Block = 4 waves, one 16(batch) x 16(hidden) tile. Wave w reduces K-quarter w:
//   w0: x @ Wih (K 0..511, + bias)   w1: x @ Wih (K 512..1023)
//   w2: h @ Whh (K 0..511)           w3: h @ Whh (K 512..1023)
// Partials meet in LDS; wave 0 sums in fixed order and applies the LSTM cell.
// 256 blocks x 4 waves = 1024 waves -> ~4/SIMD for latency hiding.
__global__ __launch_bounds__(128) void lstm_step_fused(
    const unsigned short* __restrict__ xin,   // B x H bf16 (layer input)
    const unsigned short* __restrict__ hin,   // B x H bf16 (recurrent, read parity)
    const unsigned short* __restrict__ Wih,   // NG x H bf16
    const unsigned short* __restrict__ Whh,   // NG x H bf16
    const float* __restrict__ bias,           // NG (b_ih + b_hh)
    float* __restrict__ h, float* __restrict__ c,   // B x H fp32 state (this layer)
    unsigned short* __restrict__ hout,        // B x H bf16 (write parity)
    const int* __restrict__ len,
    float* __restrict__ out,                  // B x T x H (last layer only)
    int t, int is_last)
{
  __shared__ float red[3][4][8][32];          // parts 1..3, [gate][row][lane]

  int wv   = threadIdx.x >> 5;                // 0..3 = K-quarter
  int lane = threadIdx.x & 31;
  int tile = blockIdx.x;                      // 0..255
  int m0 = (tile & 3) * 16;                   // batch tile
  int j0 = (tile >> 2) * 16;                  // hidden tile
  int col  = lane & 15;
  int mrow = (lane >> 4) << 3;                // 0 or 8

  const unsigned short* Abase =
      ((wv & 2) ? hin : xin) + (size_t)m0 * Hdim + (wv & 1) * (Hdim / 2);
  const unsigned short* Wbase = (wv & 2) ? Whh : Wih;
  const unsigned short* Wb[4];
  #pragma unroll
  for (int g = 0; g < 4; ++g)
    Wb[g] = Wbase + (size_t)(g * Hdim + j0) * Hdim + (wv & 1) * (Hdim / 2);

  v8f acc[4];
  #pragma unroll
  for (int g = 0; g < 4; ++g) {
    float bv = (wv == 0) ? bias[g * Hdim + j0 + col] : 0.0f;
    v8f a;
    #pragma unroll
    for (int r = 0; r < 8; ++r) a[r] = bv;
    acc[g] = a;
  }

  gemm_pass(Abase, Wb, acc);

  if (wv > 0) {
    #pragma unroll
    for (int g = 0; g < 4; ++g)
      #pragma unroll
      for (int r = 0; r < 8; ++r)
        red[wv - 1][g][r][lane] = acc[g][r];
  }
  __syncthreads();
  if (wv != 0) return;

  // deterministic fixed-order reduction: acc + part1 + part2 + part3
  #pragma unroll
  for (int g = 0; g < 4; ++g)
    #pragma unroll
    for (int r = 0; r < 8; ++r)
      acc[g][r] = ((acc[g][r] + red[0][g][r][lane]) + red[1][g][r][lane])
                  + red[2][g][r][lane];

  // ----- fused LSTM cell update: lane covers rows m0+mrow+r, col j0+col -----
  int j = j0 + col;
  #pragma unroll
  for (int r = 0; r < 8; ++r) {
    int b = m0 + mrow + r;
    float i_ = sigmoidf(acc[0][r]);
    float f_ = sigmoidf(acc[1][r]);
    float g_ = tanhf(acc[2][r]);
    float o_ = sigmoidf(acc[3][r]);
    size_t idx = (size_t)b * Hdim + j;
    float cold = c[idx], hold = h[idx];
    float cn = f_ * cold + i_ * g_;
    float hn = o_ * tanhf(cn);
    bool m = t < len[b];
    float h2 = m ? hn : hold;
    float c2 = m ? cn : cold;
    h[idx] = h2; c[idx] = c2; hout[idx] = f2bf(h2);
    if (is_last) out[((size_t)b * Tlen + t) * Hdim + j] = m ? hn : 0.0f;
  }
}

// hF/cF are returned in length-sorted batch order (reference keeps sorted order)
__global__ void finalize_kernel(const float* __restrict__ h, const float* __restrict__ c,
                                const int* __restrict__ order,
                                float* __restrict__ hF, float* __restrict__ cF) {
  int idx = blockIdx.x * blockDim.x + threadIdx.x;  // L*B*H
  int j = idx & (Hdim - 1);
  int i = (idx >> 10) & (Bsz - 1);
  int l = idx >> 16;                // B*H == 65536
  int src = order[i];
  size_t s = ((size_t)l * Bsz + src) * Hdim + j;
  hF[idx] = h[s]; cF[idx] = c[s];
}

// ---------------- launcher ----------------
extern "C" void kernel_launch(void* const* d_in, const int* in_sizes, int n_in,
                              void* d_out, int out_size, void* d_ws, size_t ws_size,
                              hipStream_t stream) {
  (void)in_sizes; (void)n_in; (void)out_size; (void)ws_size;
  const int*   x    = (const int*)d_in[0];
  const float* h0   = (const float*)d_in[1];
  const float* c0   = (const float*)d_in[2];
  const float* emb  = (const float*)d_in[3];
  const float* Wih  = (const float*)d_in[4];
  const float* Whh  = (const float*)d_in[5];
  const float* bih  = (const float*)d_in[6];
  const float* bhh  = (const float*)d_in[7];

  float* out = (float*)d_out;                         // (B,T,H)
  float* hF  = out + (size_t)Bsz * Tlen * Hdim;       // (L,B,H)
  float* cF  = hF + (size_t)Lnum * Bsz * Hdim;        // (L,B,H)

  char* w = (char*)d_ws;
  auto carve = [&](size_t bytes) -> void* {
    void* p = (void*)w;
    w += (bytes + 255) & ~(size_t)255;
    return p;
  };
  const size_t WH = (size_t)NG * Hdim;   // elems per matrix per layer
  const size_t BH = (size_t)Bsz * Hdim;
  unsigned short* wih_bf = (unsigned short*)carve(Lnum * WH * 2);        // 32 MB
  unsigned short* whh_bf = (unsigned short*)carve(Lnum * WH * 2);        // 32 MB
  float*          bias   = (float*)carve((size_t)Lnum * NG * 4);
  float*          hst    = (float*)carve((size_t)Lnum * BH * 4);
  float*          cst    = (float*)carve((size_t)Lnum * BH * 4);
  unsigned short* hbf2   = (unsigned short*)carve(2 * (size_t)Lnum * BH * 2); // ping-pong
  unsigned short* xbf    = (unsigned short*)carve(BH * 2);
  int*            len    = (int*)carve(256);
  int*            order  = (int*)carve(256);

  const int nW = (int)(Lnum * WH);                    // 16,777,216
  convert_weights_kernel<<<nW / 256, 256, 0, stream>>>(Wih, Whh, wih_bf, whh_bf, nW);
  bias_sum_kernel<<<(Lnum * NG) / 256, 256, 0, stream>>>(bih, bhh, bias, Lnum * NG);
  init_state_kernel<<<(Lnum * Bsz * Hdim) / 256, 256, 0, stream>>>(
      h0, c0, hst, cst, hbf2 /* parity 0 */, Lnum * Bsz * Hdim);
  lengths_order_kernel<<<1, Bsz, 0, stream>>>(x, len, order);

  for (int t = 0; t < Tlen; ++t) {
    int pr = t & 1;        // read parity (recurrent h at time t)
    int pw = pr ^ 1;       // write parity (h at time t+1 / layer input at time t)
    embed_gather_kernel<<<(Bsz * Hdim) / 256, 256, 0, stream>>>(x, emb, xbf, t);
    for (int l = 0; l < Lnum; ++l) {
      const unsigned short* xin =
          (l == 0) ? xbf : (hbf2 + ((size_t)pw * Lnum + (l - 1)) * BH);
      lstm_step_fused<<<256, 128, 0, stream>>>(
          xin,
          hbf2 + ((size_t)pr * Lnum + l) * BH,
          wih_bf + (size_t)l * WH, whh_bf + (size_t)l * WH,
          bias + (size_t)l * NG,
          hst + (size_t)l * BH, cst + (size_t)l * BH,
          hbf2 + ((size_t)pw * Lnum + l) * BH,
          len, out, t, (l == Lnum - 1) ? 1 : 0);
    }
  }
  finalize_kernel<<<(Lnum * Bsz * Hdim) / 256, 256, 0, stream>>>(hst, cst, order, hF, cF);
}
